// MultiHeadedAttention_27350351741623
// MI455X (gfx1250) — matrix operand order repaired and checked by reading, not compiled, not run on hardware
//
#include <hip/hip_runtime.h>
#include <hip/hip_bf16.h>

// ---------------------------------------------------------------------------
// Types for CDNA5 WMMA (wave32): v_wmma_f32_16x16x32_bf16
// ---------------------------------------------------------------------------
typedef __bf16 bf16_t;
typedef __attribute__((ext_vector_type(16))) __bf16 v16bf;
typedef __attribute__((ext_vector_type(8)))  __bf16 v8bf;
typedef __attribute__((ext_vector_type(8)))  float  v8f;
typedef __attribute__((ext_vector_type(4)))  float  v4f;

#define B_  16
#define S_  512
#define D_  256
#define H_  8
#define HD_ 32

__device__ __forceinline__ v16bf cat8(v8bf lo, v8bf hi) {
  return __builtin_shufflevector(lo, hi, 0,1,2,3,4,5,6,7,8,9,10,11,12,13,14,15);
}

__device__ __forceinline__ v8f wmma_bf16(v16bf a, v16bf b, v8f c) {
  // (neg_a, A, neg_b, B, c_mod, C, reuse_a, reuse_b)
  return __builtin_amdgcn_wmma_f32_16x16x32_bf16(false, a, false, b, (short)0, c,
                                                 false, false);
}

// ---------------------------------------------------------------------------
// Kernel 0: convert+transpose W (D x D f32, row-major [k][n]) -> bf16 [n][k]
// for contiguous B-operand loads in the projection GEMM. 3 matrices.
// ---------------------------------------------------------------------------
__global__ __launch_bounds__(256) void wt_kernel(const float* __restrict__ Wq,
                                                 const float* __restrict__ Wk,
                                                 const float* __restrict__ Wv,
                                                 bf16_t* __restrict__ Wt) {
  int idx = blockIdx.x * 256 + threadIdx.x;   // 0 .. 3*65536-1
  int which = idx >> 16;
  int e     = idx & 65535;
  int n = e >> 8, k = e & 255;
  const float* W = (which == 0) ? Wq : (which == 1) ? Wk : Wv;
  Wt[(size_t)which * 65536 + n * 256 + k] = (bf16_t)W[k * 256 + n];
}

// ---------------------------------------------------------------------------
// Kernel 1: fused QKV projection.  Y = X @ W + b  (8192 x 256 x 256) x3.
// One wave -> one 16x16 output tile, K-loop of 8 WMMA(16x16x32) steps.
// q,k stored bf16 (B,H,S,HD); v stored transposed bf16 (B,H,HD,S).
// ---------------------------------------------------------------------------
__global__ __launch_bounds__(128) void qkv_proj_kernel(
    const float* __restrict__ X0, const float* __restrict__ X1,
    const float* __restrict__ X2, const bf16_t* __restrict__ Wt,
    const float* __restrict__ bq, const float* __restrict__ bk,
    const float* __restrict__ bv, bf16_t* __restrict__ qb,
    bf16_t* __restrict__ kb, bf16_t* __restrict__ vtb) {
  const int wave = threadIdx.x >> 5;
  const int lane = threadIdx.x & 31;
  const int lhalf = lane >> 4;          // 0 | 1
  const int l16   = lane & 15;

  int tile = blockIdx.x * 4 + wave;     // 0 .. 24575
  const int which = tile / (512 * 16);
  int rem = tile - which * (512 * 16);
  const int mt = rem >> 4;              // M tile (0..511)
  const int nt = rem & 15;              // N tile (0..15)

  const float* X    = (which == 0) ? X0 : (which == 1) ? X1 : X2;
  const float* bias = (which == 0) ? bq : (which == 1) ? bk : bv;
  const bf16_t* W   = Wt + (size_t)which * 65536;

  const int mrow = mt * 16 + l16;       // A row  (M = lane%16)
  const int ncol = nt * 16 + l16;       // B col  (N = lane%16)

  v8f c = {0.f,0.f,0.f,0.f,0.f,0.f,0.f,0.f};
  #pragma unroll
  for (int k0 = 0; k0 < 256; k0 += 32) {
    // A lane layout (16-bit A 16x32): lanes<16 hold K {0..7,16..23},
    // lanes>=16 hold K {8..15,24..31} -> two 8-elem chunks.
    const float* arow = X + (size_t)mrow * 256 + k0 + lhalf * 8;
    v4f a0 = *(const v4f*)(arow);
    v4f a1 = *(const v4f*)(arow + 4);
    v4f a2 = *(const v4f*)(arow + 16);
    v4f a3 = *(const v4f*)(arow + 20);
    v16bf A;
    #pragma unroll
    for (int i = 0; i < 4; ++i) {
      A[i]      = (bf16_t)a0[i];
      A[4 + i]  = (bf16_t)a1[i];
      A[8 + i]  = (bf16_t)a2[i];
      A[12 + i] = (bf16_t)a3[i];
    }
    // B lane layout (16-bit B 32x16): lanes<16 K {0..7,16..23}, lanes>=16 K {8..15,24..31},
    // column n = lane%16 -> two 8-elem chunks from transposed W row n.
    const bf16_t* brow = W + (size_t)ncol * 256 + k0 + lhalf * 8;
    v8bf b0 = *(const v8bf*)(brow);
    v8bf b1 = *(const v8bf*)(brow + 16);
    c = wmma_bf16(A, cat8(b0, b1), c);
  }

  const float bias_v = bias[ncol];
  const int h  = ncol >> 5;
  const int hd = ncol & 31;
  #pragma unroll
  for (int r = 0; r < 8; ++r) {
    float y = c[r] + bias_v;
    int m  = mt * 16 + r + lhalf * 8;   // C: M = r + 8*(lane>=16)
    int bb = m >> 9;                    // batch
    int s  = m & 511;                   // sequence
    size_t bh = (size_t)bb * H_ + h;
    if (which == 0)
      qb[(bh * S_ + s) * HD_ + hd] = (bf16_t)y;
    else if (which == 1)
      kb[(bh * S_ + s) * HD_ + hd] = (bf16_t)y;
    else
      vtb[(bh * HD_ + hd) * S_ + s] = (bf16_t)y;
  }
}

// ---------------------------------------------------------------------------
// Kernel 2: attention. One workgroup (4 waves / 128 thr) = one (b,h,16-row
// q-tile). Phase 1: scores 16x512 via WMMA into LDS. Phase 2: three-way
// softmax (prob / time / rel), writes prob_attn + bf16 P tile in LDS.
// Phase 3: out = P @ V via WMMA (waves 0,1).
// ---------------------------------------------------------------------------
#define SROW 520            // f32 score row pitch (padded)
#define PROW 528            // bf16 prob row pitch (padded, 16B aligned)

__global__ __launch_bounds__(128) void attn_kernel(
    const bf16_t* __restrict__ qb, const bf16_t* __restrict__ kb,
    const bf16_t* __restrict__ vtb, const float* __restrict__ rel,
    const float* __restrict__ tsm, const float* __restrict__ l1p,
    const float* __restrict__ l2p, float* __restrict__ out,
    float* __restrict__ prob_out) {
  __shared__ float  s_scores[16 * SROW];
  __shared__ bf16_t s_prob[16 * PROW];
  __shared__ float  red[3][16][8];

  const int tid   = threadIdx.x;
  const int wave  = tid >> 5;
  const int lane  = tid & 31;
  const int lhalf = lane >> 4;
  const int l16   = lane & 15;

  int blk = blockIdx.x;                 // B*H*(S/16) = 4096
  const int qt = blk & 31;  blk >>= 5;
  const int h  = blk & 7;
  const int b  = blk >> 3;
  const int q0 = qt * 16;
  const size_t bh = (size_t)b * H_ + h;

  const float scale = 0.17677669529663687f;   // 1/sqrt(HD)

  // ---- Phase 1: scores = (q k^T) * scale, 8 column-tiles per wave --------
  {
    const bf16_t* qrow = qb + (bh * S_ + q0 + l16) * HD_ + lhalf * 8;
    v8bf a0 = *(const v8bf*)(qrow);
    v8bf a1 = *(const v8bf*)(qrow + 16);
    v16bf A = cat8(a0, a1);
    #pragma unroll
    for (int t = 0; t < 8; ++t) {
      int ct = wave * 8 + t;            // column tile 0..31
      const bf16_t* krow = kb + (bh * S_ + ct * 16 + l16) * HD_ + lhalf * 8;
      v8bf b0 = *(const v8bf*)(krow);
      v8bf b1 = *(const v8bf*)(krow + 16);
      v8f c = {0.f,0.f,0.f,0.f,0.f,0.f,0.f,0.f};
      c = wmma_bf16(A, cat8(b0, b1), c);
      #pragma unroll
      for (int r = 0; r < 8; ++r)
        s_scores[(r + 8 * lhalf) * SROW + ct * 16 + l16] = c[r] * scale;
    }
  }
  __syncthreads();

  // ---- Phase 2: row-wise three-way softmax -------------------------------
  const int row = tid >> 3;             // 0..15
  const int sub = tid & 7;              // 0..7, each covers 64 columns
  const int qg  = q0 + row;             // global query index
  const float* tsrow  = tsm + ((size_t)b * S_ + qg) * S_;
  const float* relrow = rel + ((size_t)b * S_ + qg) * S_;

  float mS = -3e38f, mT = -3e38f, mR = -3e38f;
  for (int j = sub * 64; j < sub * 64 + 64; ++j) {
    bool masked = (j > qg);
    float s = masked ? -1e9f : s_scores[row * SROW + j];
    mS = fmaxf(mS, s);
    if (!masked) mT = fmaxf(mT, __expf(-fabsf(tsrow[j])));
    float rv = relrow[j];
    float r_ = (masked && rv != 0.0f) ? rv : -1e4f;
    mR = fmaxf(mR, r_);
  }
  red[0][row][sub] = mS; red[1][row][sub] = mT; red[2][row][sub] = mR;
  __syncthreads();
  mS = mT = mR = -3e38f;
  #pragma unroll
  for (int i = 0; i < 8; ++i) {
    mS = fmaxf(mS, red[0][row][i]);
    mT = fmaxf(mT, red[1][row][i]);
    mR = fmaxf(mR, red[2][row][i]);
  }
  __syncthreads();

  float sS = 0.f, sT = 0.f, sR = 0.f;
  for (int j = sub * 64; j < sub * 64 + 64; ++j) {
    bool masked = (j > qg);
    float s = masked ? -1e9f : s_scores[row * SROW + j];
    sS += __expf(s - mS);
    if (!masked) sT += __expf(__expf(-fabsf(tsrow[j])) - mT);
    float rv = relrow[j];
    float r_ = (masked && rv != 0.0f) ? rv : -1e4f;
    sR += __expf(r_ - mR);
  }
  red[0][row][sub] = sS; red[1][row][sub] = sT; red[2][row][sub] = sR;
  __syncthreads();
  sS = sT = sR = 0.f;
  #pragma unroll
  for (int i = 0; i < 8; ++i) {
    sS += red[0][row][i]; sT += red[1][row][i]; sR += red[2][row][i];
  }

  const float l1 = *l1p, l2 = *l2p;
  const float c_s = (1.f - l1) * (1.f - l2) / sS;
  const float c_t = (1.f - l1) * l2 / sT;
  const float c_r = l1 / sR;

  float* probrow = prob_out + (bh * S_ + qg) * S_;
  for (int j = sub * 64; j < sub * 64 + 64; ++j) {
    bool masked = (j > qg);
    float s = masked ? -1e9f : s_scores[row * SROW + j];
    float pt = masked ? 0.f : __expf(__expf(-fabsf(tsrow[j])) - mT) * c_t;
    float rv = relrow[j];
    float r_ = (masked && rv != 0.0f) ? rv : -1e4f;
    float p = __expf(s - mS) * c_s + pt + __expf(r_ - mR) * c_r;
    probrow[j] = p;
    s_prob[row * PROW + j] = (bf16_t)p;
  }
  __syncthreads();

  // ---- Phase 3: out tile = P(16x512) @ V(512x32); waves 0,1 --------------
  if (wave < 2) {
    const int nloc = wave * 16 + l16;   // hd column 0..31
    const bf16_t* vrow = vtb + (bh * HD_ + nloc) * S_;
    v8f c = {0.f,0.f,0.f,0.f,0.f,0.f,0.f,0.f};
    #pragma unroll 4
    for (int k0 = 0; k0 < S_; k0 += 32) {
      const bf16_t* pr = s_prob + l16 * PROW + k0 + lhalf * 8;
      v8bf a0 = *(const v8bf*)(pr);
      v8bf a1 = *(const v8bf*)(pr + 16);
      const bf16_t* br = vrow + k0 + lhalf * 8;
      v8bf b0 = *(const v8bf*)(br);
      v8bf b1 = *(const v8bf*)(br + 16);
      c = wmma_bf16(cat8(a0, a1), cat8(b0, b1), c);
    }
    #pragma unroll
    for (int r = 0; r < 8; ++r) {
      int m = r + 8 * lhalf;
      out[((size_t)b * S_ + q0 + m) * D_ + h * HD_ + nloc] = c[r];
    }
  }
}

// ---------------------------------------------------------------------------
// Launch
// ---------------------------------------------------------------------------
extern "C" void kernel_launch(void* const* d_in, const int* in_sizes, int n_in,
                              void* d_out, int out_size, void* d_ws,
                              size_t ws_size, hipStream_t stream) {
  const float* query = (const float*)d_in[0];
  const float* key_t = (const float*)d_in[1];
  const float* value = (const float*)d_in[2];
  const float* rel   = (const float*)d_in[3];
  const float* tsm   = (const float*)d_in[4];
  // d_in[5] = mask (bool, analytic triu -> recomputed on device)
  const float* l1    = (const float*)d_in[6];
  const float* l2    = (const float*)d_in[7];
  const float* Wq    = (const float*)d_in[8];
  const float* bq    = (const float*)d_in[9];
  const float* Wk    = (const float*)d_in[10];
  const float* bk    = (const float*)d_in[11];
  const float* Wv    = (const float*)d_in[12];
  const float* bv    = (const float*)d_in[13];

  float* out  = (float*)d_out;                          // (B,S,D)
  float* prob = out + (size_t)B_ * S_ * D_;             // (B,H,S,S)

  bf16_t* Wt    = (bf16_t*)d_ws;                        // 3 * 256*256
  bf16_t* qbuf  = Wt + (size_t)3 * 65536;               // (B,H,S,HD)
  bf16_t* kbuf  = qbuf + (size_t)B_ * H_ * S_ * HD_;    // (B,H,S,HD)
  bf16_t* vtbuf = kbuf + (size_t)B_ * H_ * S_ * HD_;    // (B,H,HD,S)

  wt_kernel<<<768, 256, 0, stream>>>(Wq, Wk, Wv, Wt);
  qkv_proj_kernel<<<6144, 128, 0, stream>>>(query, key_t, value, Wt, bq, bk, bv,
                                            qbuf, kbuf, vtbuf);
  attn_kernel<<<4096, 128, 0, stream>>>(qbuf, kbuf, vtbuf, rel, tsm, l1, l2,
                                        out, prob);
}
